// StepwiseMonotonicAttention_74483322847558
// MI455X (gfx1250) — compile-verified, matrix operand order checked by reading, not run on hardware
//
#include <hip/hip_runtime.h>
#include <hip/hip_bf16.h>

typedef __attribute__((ext_vector_type(16))) __bf16 v16bf;
typedef __attribute__((ext_vector_type(8)))  float  v8f;

#define B_ 32
#define T_ 4096
#define E_ 512
#define A_ 128
#define KC_ 17                    // 16 chunks of value-K + 1 chunk holding the 31-tap conv
#define LDSB_ELEMS (KC_*8*32*16)  // 69632 bf16
#define LDSB_BYTES (LDSB_ELEMS*2) // 139264 B

#if __has_builtin(__builtin_amdgcn_tanhf)
#define TANHF(x) __builtin_amdgcn_tanhf(x)
#elif __has_builtin(__builtin_amdgcn_tanh_f32)
#define TANHF(x) __builtin_amdgcn_tanh_f32(x)
#else
#define TANHF(x) tanhf(x)
#endif

// ---------------- workspace layout (bytes) ----------------
#define WS_WVS   0            // bf16 swizzled [Wv ; W2] : 139264
#define WS_BASE  139264       // float [B][A]            : 16384
#define WS_NV    155648       // float [A]               : 512
#define WS_W2    156160       // float [31][A]           : 15872
#define WS_P     172032       // float [B][T]            : 524288
#define WS_PART  696320       // float [B][16][E]        : 1048576

// ================= P1: small weight prep =================
__global__ void smha_prep(const float* __restrict__ query,
                          const float* __restrict__ Wq,
                          const float* __restrict__ convk,  // [31][1][32]
                          const float* __restrict__ Wl,     // [32][128]
                          const float* __restrict__ att_v,
                          const float* __restrict__ att_g,
                          const float* __restrict__ att_b,
                          const float* __restrict__ bv,
                          float* __restrict__ base,
                          float* __restrict__ nv,
                          float* __restrict__ W2) {
  int tid = threadIdx.x;
  if (tid < A_) {
    float ss = 0.f;
    for (int i = 0; i < A_; ++i) { float v = att_v[i]; ss += v * v; }
    nv[tid] = att_g[0] * att_v[tid] * rsqrtf(ss);
  }
  for (int idx = tid; idx < 31 * A_; idx += 256) {
    int k = idx / A_, a = idx % A_;
    float s = 0.f;
    for (int c = 0; c < 32; ++c) s += convk[k * 32 + c] * Wl[c * A_ + a];
    W2[idx] = s;
  }
  for (int idx = tid; idx < B_ * A_; idx += 256) {
    int b = idx / A_, a = idx % A_;
    const float* q = query + b * 512;
    float s = 0.f;
    for (int e = 0; e < 512; ++e) s += q[e] * Wq[e * A_ + a];
    base[idx] = s + bv[a] + att_b[a];
  }
}

// ====== P2: swizzle [Wv ; W2 ; 0] into per-lane bf16 B-fragment layout ======
// element index = ((kc*8+nt)*32 + lane)*16 + j ; n = nt*16 + (lane&15)
// k = kc*32 + (lane>>4)*8 + j + (j&8)
__global__ void smha_swz(const float* __restrict__ Wv,
                         const float* __restrict__ W2,
                         __bf16* __restrict__ WvS) {
  int idx = blockIdx.x * 256 + threadIdx.x;
  if (idx >= LDSB_ELEMS) return;
  int j    = idx & 15;
  int lane = (idx >> 4) & 31;
  int cn   = idx >> 9;            // kc*8 + nt
  int nt   = cn & 7, kc = cn >> 3;
  int n = nt * 16 + (lane & 15);
  int k = kc * 32 + (lane >> 4) * 8 + j + (j & 8);
  float v;
  if (k < 512) v = Wv[k * A_ + n];
  else { int kk = k - 512; v = (kk < 31) ? W2[kk * A_ + n] : 0.f; }
  WvS[idx] = (__bf16)v;
}

// ================= MAIN: fused GEMM + conv + energy -> p =================
__global__ __launch_bounds__(256)
void smha_main(const float* __restrict__ value,
               const int*   __restrict__ vlen,
               const float* __restrict__ accum,
               const __bf16* __restrict__ WvS,
               const float* __restrict__ base,
               const float* __restrict__ nv,
               const float* __restrict__ sbias,
               float* __restrict__ p_out) {
  extern __shared__ char smem[];
  v16bf* Bs = (v16bf*)smem;
  {
    const uint4* src = (const uint4*)WvS;
    uint4* dst = (uint4*)smem;
    for (int i = threadIdx.x; i < LDSB_BYTES / 16; i += 256) dst[i] = src[i];
  }
  __syncthreads();

  const int wg   = blockIdx.x;          // 1024 WGs, 128 rows each
  const int row0 = wg * 128;
  const int b    = row0 / T_;
  const int t0   = row0 % T_;
  const int lane = threadIdx.x & 31;
  const int w    = threadIdx.x >> 5;    // wave 0..7 -> 16 rows each
  const int half = lane >> 4;
  const int mrow = lane & 15;
  const int rloc = w * 16 + mrow;
  const int t    = t0 + rloc;
  const float* arow  = value + (size_t)(b * T_ + t) * E_;
  const float* acrow = accum + (size_t)b * T_;

  v8f acc[8];
#pragma unroll
  for (int nt = 0; nt < 8; ++nt) acc[nt] = (v8f){0.f,0.f,0.f,0.f,0.f,0.f,0.f,0.f};

  // block-uniform: can the whole 128-row tile read accum[t-15 .. t+16] unguarded?
  const bool interior = (t0 >= 15) && (t0 <= T_ - 144);

  for (int kc = 0; kc < KC_; ++kc) {
    v16bf af;
    if (kc < 16) {
      const float4* pa = (const float4*)(arow + kc * 32 + half * 8);
      float4 a0 = pa[0], a1 = pa[1];      // K = k0 + half*8 + 0..7
      float4 a2 = pa[4], a3 = pa[5];      // K = k0 + 16 + half*8 + 0..7
      af[0]=(__bf16)a0.x; af[1]=(__bf16)a0.y; af[2]=(__bf16)a0.z; af[3]=(__bf16)a0.w;
      af[4]=(__bf16)a1.x; af[5]=(__bf16)a1.y; af[6]=(__bf16)a1.z; af[7]=(__bf16)a1.w;
      af[8]=(__bf16)a2.x; af[9]=(__bf16)a2.y; af[10]=(__bf16)a2.z; af[11]=(__bf16)a2.w;
      af[12]=(__bf16)a3.x; af[13]=(__bf16)a3.y; af[14]=(__bf16)a3.z; af[15]=(__bf16)a3.w;
    } else if (interior) {
      // conv chunk fast path: koff==31 reads a live value but hits the zero B row
#pragma unroll
      for (int j = 0; j < 16; ++j) {
        int koff = half * 8 + j + (j & 8);
        af[j] = (__bf16)acrow[t + koff - 15];
      }
    } else {
      // boundary blocks: branchless clamp + select (SAME zero padding)
#pragma unroll
      for (int j = 0; j < 16; ++j) {
        int koff = half * 8 + j + (j & 8);
        int ti   = t + koff - 15;
        int tic  = min(max(ti, 0), T_ - 1);
        float v  = acrow[tic];
        af[j] = (__bf16)(((ti == tic) && (koff < 31)) ? v : 0.f);
      }
    }
    const v16bf* bp = Bs + kc * 8 * 32;
#pragma unroll
    for (int nt = 0; nt < 8; ++nt) {
      v16bf bfrag = bp[nt * 32 + lane];
      acc[nt] = __builtin_amdgcn_wmma_f32_16x16x32_bf16(
          false, af, false, bfrag, (short)0, acc[nt], false, false);
    }
  }

  // epilogue: energy -> sigmoid -> p
  float nvv[8], bs[8];
  const float* baseb = base + b * A_;
#pragma unroll
  for (int nt = 0; nt < 8; ++nt) {
    int n = nt * 16 + mrow;
    nvv[nt] = nv[n];
    bs[nt]  = baseb[n];
  }
  const float sb  = sbias[0];
  const int   len = vlen[b];
#pragma unroll
  for (int r = 0; r < 8; ++r) {
    float s = 0.f;
#pragma unroll
    for (int nt = 0; nt < 8; ++nt) s += nvv[nt] * TANHF(acc[nt][r] + bs[nt]);
    s += __shfl_xor(s, 1); s += __shfl_xor(s, 2);
    s += __shfl_xor(s, 4); s += __shfl_xor(s, 8);   // reduce over 16-lane N group
    int tt = t0 + w * 16 + r + 8 * half;            // C/D layout: M = r + 8*half
    float e = s + sb;
    if (tt >= len) e += -1e9f;
    float p = 1.f / (1.f + __expf(-e));
    if (mrow == 0) p_out[(size_t)b * T_ + tt] = p;
  }
}

// ================= AW: monotonic update =================
__global__ void smha_aw(const float* __restrict__ prev,
                        const float* __restrict__ p,
                        float* __restrict__ aw) {
  int i = blockIdx.x * 256 + threadIdx.x;
  if (i >= B_ * T_) return;
  int tt = i & (T_ - 1);
  float v = prev[i] * p[i];
  if (tt > 0) v += prev[i - 1] * (1.f - p[i - 1]);
  aw[i] = v;
}

// ================= CTX: partial context sums (deterministic) =================
__global__ void smha_ctx(const float* __restrict__ value,
                         const float* __restrict__ aw,
                         float* __restrict__ partial) {
  __shared__ float awl[256];
  int b = blockIdx.z, tc = blockIdx.y;
  int e = blockIdx.x * 256 + threadIdx.x;
  awl[threadIdx.x] = aw[(size_t)b * T_ + tc * 256 + threadIdx.x];
  __syncthreads();
  const float* vb = value + ((size_t)b * T_ + tc * 256) * E_ + e;
  float s = 0.f;
#pragma unroll 4
  for (int tt = 0; tt < 256; ++tt) s += awl[tt] * vb[(size_t)tt * E_];
  partial[(size_t)(b * 16 + tc) * E_ + e] = s;
}

__global__ void smha_red(const float* __restrict__ partial,
                         float* __restrict__ out) {
  int i = blockIdx.x * 256 + threadIdx.x;
  if (i >= B_ * E_) return;
  int b = i / E_, e = i % E_;
  float s = 0.f;
#pragma unroll
  for (int tc = 0; tc < 16; ++tc) s += partial[(size_t)(b * 16 + tc) * E_ + e];
  out[i] = s;
}

extern "C" void kernel_launch(void* const* d_in, const int* in_sizes, int n_in,
                              void* d_out, int out_size, void* d_ws, size_t ws_size,
                              hipStream_t stream) {
  const float* value  = (const float*)d_in[0];
  const int*   vlen   = (const int*)  d_in[1];
  const float* query  = (const float*)d_in[2];
  const float* accum  = (const float*)d_in[3];
  const float* prev   = (const float*)d_in[4];
  const float* Wv     = (const float*)d_in[5];
  const float* bv     = (const float*)d_in[6];
  const float* Wq     = (const float*)d_in[7];
  const float* convk  = (const float*)d_in[8];
  const float* Wl     = (const float*)d_in[9];
  const float* att_v  = (const float*)d_in[10];
  const float* att_g  = (const float*)d_in[11];
  const float* att_b  = (const float*)d_in[12];
  const float* sbias  = (const float*)d_in[13];

  char* ws = (char*)d_ws;
  __bf16* WvS   = (__bf16*)(ws + WS_WVS);
  float*  base  = (float*) (ws + WS_BASE);
  float*  nv    = (float*) (ws + WS_NV);
  float*  W2    = (float*) (ws + WS_W2);
  float*  p     = (float*) (ws + WS_P);
  float*  part  = (float*) (ws + WS_PART);

  float* attn_c = (float*)d_out;           // [B][E]
  float* aw     = (float*)d_out + B_ * E_; // [B][T]

  smha_prep<<<1, 256, 0, stream>>>(query, Wq, convk, Wl, att_v, att_g, att_b, bv,
                                   base, nv, W2);
  smha_swz<<<(LDSB_ELEMS + 255) / 256, 256, 0, stream>>>(Wv, W2, WvS);
  smha_main<<<(B_ * T_) / 128, 256, LDSB_BYTES, stream>>>(value, vlen, accum, WvS,
                                                          base, nv, sbias, p);
  smha_aw<<<(B_ * T_) / 256, 256, 0, stream>>>(prev, p, aw);
  smha_ctx<<<dim3(E_ / 256, 16, B_), 256, 0, stream>>>(value, aw, part);
  smha_red<<<(B_ * E_) / 256, 256, 0, stream>>>(part, attn_c);
}
